// multi_head_attention_30666066494217
// MI455X (gfx1250) — matrix operand order repaired and checked by reading, not compiled, not run on hardware
//
#include <hip/hip_runtime.h>
#include <hip/hip_bf16.h>

// ---------------------------------------------------------------------------
// Types for CDNA5 WMMA (wave32): bf16 16x16x32, f32 accumulate.
// ---------------------------------------------------------------------------
typedef __bf16 bh;
typedef __attribute__((ext_vector_type(8)))  bh    bh8;
typedef __attribute__((ext_vector_type(16))) bh    bh16;
typedef __attribute__((ext_vector_type(8)))  float vfloat8;
typedef __attribute__((ext_vector_type(4)))  float vfloat4;

__device__ __forceinline__ vfloat8 zero8() {
    vfloat8 z;
#pragma unroll
    for (int i = 0; i < 8; ++i) z[i] = 0.0f;
    return z;
}

// D = A x B + C  (16x16 f32 out, K=32 bf16)
__device__ __forceinline__ vfloat8 wmma_bf16(bh16 a, bh16 b, vfloat8 c) {
    return __builtin_amdgcn_wmma_f32_16x16x32_bf16(
        /*neg_a=*/false, a, /*neg_b=*/false, b,
        /*c_mod=*/(short)0, c, /*reuse_a=*/false, /*reuse_b=*/false);
}

// ---------------------------------------------------------------------------
// Operand loaders. Source matrices row-major, K contiguous.
// A 16x32 bf16 (ISA 7.12.2): lane m=lane&15 holds row m;
//   lanes 0-15: K 0..7 | 16..23 ; lanes 16-31: K 8..15 | 24..31
// B 32x16 bf16: lane n=lane&15 holds column n (= source row n);
//   lanes 0-15: K 0..7 | 16..23 ; lanes 16-31: K 8..15 | 24..31
// ---------------------------------------------------------------------------
__device__ __forceinline__ bh16 load_a_bf16(const bh* base, int ld, int row0, int k0) {
    const int lane = threadIdx.x & 31;
    const int m  = lane & 15;
    const int kh = (lane >> 4) << 3;          // 0 or 8
    const bh* p = base + (size_t)(row0 + m) * ld + k0 + kh;
    bh8 lo = *(const bh8*)p;                  // K kh..kh+7
    bh8 hi = *(const bh8*)(p + 16);           // K kh+16..kh+23
    return __builtin_shufflevector(lo, hi, 0,1,2,3,4,5,6,7,8,9,10,11,12,13,14,15);
}

__device__ __forceinline__ bh16 load_b_bf16(const bh* base, int ld, int row0, int k0) {
    const int lane = threadIdx.x & 31;
    const int n  = lane & 15;
    const int ko = (lane >> 4) << 3;          // 0 or 8
    const bh* p = base + (size_t)(row0 + n) * ld + k0 + ko;
    bh8 lo = *(const bh8*)p;
    bh8 hi = *(const bh8*)(p + 16);
    return __builtin_shufflevector(lo, hi, 0,1,2,3,4,5,6,7,8,9,10,11,12,13,14,15);
}

// ---------------------------------------------------------------------------
// Kernel 0: f32 -> bf16 conversion (memory bound, one pass).
// ---------------------------------------------------------------------------
__global__ __launch_bounds__(256) void cvt_bf16_kernel(
    const float* __restrict__ in, bh* __restrict__ out, int n)
{
    const int i = (blockIdx.x * 256 + threadIdx.x) * 8;
    if (i + 8 <= n) {
        vfloat4 f0 = *(const vfloat4*)(in + i);
        vfloat4 f1 = *(const vfloat4*)(in + i + 4);
        bh8 v;
#pragma unroll
        for (int j = 0; j < 4; ++j) { v[j] = (bh)f0[j]; v[4 + j] = (bh)f1[j]; }
        *(bh8*)(out + i) = v;
    }
}

// ---------------------------------------------------------------------------
// Kernel 1: projection  Out = (X @ W^T + bias) * outScale   (8192x1024x1024)
// X, W already bf16. Block = 256 threads (8 waves); wave tile = 32 x 128
// (2 A tiles x 8 B tiles -> 16 WMMAs per K-step, 20 b128 loads: 1.25 loads/wmma).
// transposeOut=1 stores Out^T (1024 x 8192) for V so the attention P@V
// B-operand is K-contiguous.
// ---------------------------------------------------------------------------
__global__ __launch_bounds__(256) void proj_gemm_kernel(
    const bh* __restrict__ X, const bh* __restrict__ W,
    const float* __restrict__ bias, bh* __restrict__ out,
    float outScale, int transposeOut)
{
    const int w    = threadIdx.x >> 5;
    const int lane = threadIdx.x & 31;
    const int m0 = blockIdx.x * 256 + w * 32;
    const int n0 = blockIdx.y * 128;

    vfloat8 acc[2][8];
#pragma unroll
    for (int h = 0; h < 2; ++h)
#pragma unroll
        for (int t = 0; t < 8; ++t) acc[h][t] = zero8();

    for (int k0 = 0; k0 < 1024; k0 += 32) {
        bh16 a0 = load_a_bf16(X, 1024, m0,      k0);
        bh16 a1 = load_a_bf16(X, 1024, m0 + 16, k0);
#pragma unroll
        for (int t = 0; t < 8; ++t) {
            bh16 b = load_b_bf16(W, 1024, n0 + t * 16, k0);
            acc[0][t] = wmma_bf16(a0, b, acc[0][t]);
            acc[1][t] = wmma_bf16(a1, b, acc[1][t]);
        }
    }

    // C layout: lane n holds column n; VGPR r holds row mb+r, mb=(lane>>4)*8.
    const int n  = lane & 15;
    const int mb = (lane >> 4) << 3;
#pragma unroll
    for (int h = 0; h < 2; ++h) {
        const int mrow = m0 + h * 16 + mb;
        if (!transposeOut) {
#pragma unroll
            for (int t = 0; t < 8; ++t) {
                float bv = bias[n0 + t * 16 + n];
#pragma unroll
                for (int r = 0; r < 8; ++r)
                    out[(size_t)(mrow + r) * 1024 + n0 + t * 16 + n] =
                        (bh)((acc[h][t][r] + bv) * outScale);
            }
        } else {
#pragma unroll
            for (int t = 0; t < 8; ++t) {
                float bv = bias[n0 + t * 16 + n];
                bh8 v;
#pragma unroll
                for (int r = 0; r < 8; ++r) v[r] = (bh)((acc[h][t][r] + bv) * outScale);
                *(bh8*)(out + (size_t)(n0 + t * 16 + n) * 8192 + mrow) = v; // 16B store
            }
        }
    }
}

// ---------------------------------------------------------------------------
// Kernel 2: one-pass flash attention with KV-split=2 (for machine fill).
//   Q: 8192x1024 bf16 (pre-scaled by 1/32), K: 8192x1024 bf16,
//   Vt: 1024x8192 bf16, Opart: [2][8192][1024] f32 (UNnormalized),
//   Mst/Lst: [2][8192] f32 row stats.
// Block = 512 threads = 16 waves, Br = 64 query rows, Bc = 32 keys/iter,
// 128 iterations per block (half the keys). Grid = (128 q-blocks, 2 kv-halves).
//   S phase : wave (qt,kt,dh) -> one 16x16 S tile partial over a 512-wide d
//             half; Q operands persist in 128 VGPRs for the whole kernel.
//   softmax : 512 threads, 8 lanes/row, shfl_xor reductions.
//   O phase : wave (qt,dq) owns 16 rows x 256 cols of O (128 VGPRs of f32);
//             alpha-rescale + 16 WMMAs of P @ Vt per iteration.
// Next KV block is prefetched (global_prefetch_b8) one iteration ahead.
// ---------------------------------------------------------------------------
__global__ __launch_bounds__(512) void flash_attn_kernel(
    const bh* __restrict__ Q, const bh* __restrict__ K,
    const bh* __restrict__ Vt, float* __restrict__ Opart,
    float* __restrict__ Mst, float* __restrict__ Lst)
{
    __shared__ float Spart[2][64][32];          // d-half partials of S block
    __shared__ __align__(16) bh Pc[64][32];     // softmaxed probabilities (bf16)
    __shared__ float sm[64], sl[64], salpha[64];

    const int t    = threadIdx.x;
    const int w    = t >> 5;
    const int lane = t & 31;
    const int qt = w >> 2;          // 0..3 : query 16-row tile
    const int kt = (w >> 1) & 1;    // 0..1 : key 16-col tile (S phase)
    const int dh = w & 1;           // 0..1 : 512-wide d half   (S phase)
    const int dq = w & 3;           // 0..3 : 256-wide d chunk  (O phase)
    const int q0 = blockIdx.x * 64;
    const int s  = blockIdx.y;      // kv split index
    const int kbBeg = s * 128, kbEnd = kbBeg + 128;

    if (t < 64) { sm[t] = -3.0e38f; sl[t] = 0.0f; }

    // Preload this wave's Q operands (16 A tiles = 128 VGPRs).
    bh16 qa[16];
#pragma unroll
    for (int ks = 0; ks < 16; ++ks)
        qa[ks] = load_a_bf16(Q, 1024, q0 + qt * 16, dh * 512 + ks * 32);

    vfloat8 o[16];
#pragma unroll
    for (int i = 0; i < 16; ++i) o[i] = zero8();

    __syncthreads();

    for (int kb = kbBeg; kb < kbEnd; ++kb) {
        const int key0 = kb * 32;

        // Prefetch next 32-key block: K rows are 64KB contiguous (1 line/thread),
        // Vt needs 1024 row-segments (2 lines/thread).
        if (kb + 1 < kbEnd) {
            const bh* kn = K + (size_t)(key0 + 32) * 1024;
            __builtin_prefetch(kn + (size_t)t * 64, 0, 3);
            const bh* vn = Vt + (size_t)t * 8192 + key0 + 32;
            __builtin_prefetch(vn, 0, 3);
            __builtin_prefetch(vn + (size_t)512 * 8192, 0, 3);
        }

        // ---- Phase A: S partial (16 WMMAs over this wave's d-half) ----
        {
            vfloat8 sacc = zero8();
#pragma unroll
            for (int ks = 0; ks < 16; ++ks) {
                bh16 b = load_b_bf16(K, 1024, key0 + kt * 16, dh * 512 + ks * 32);
                sacc = wmma_bf16(qa[ks], b, sacc);
            }
            const int n  = lane & 15;
            const int mb = (lane >> 4) << 3;
#pragma unroll
            for (int r = 0; r < 8; ++r)
                Spart[dh][qt * 16 + mb + r][kt * 16 + n] = sacc[r];
        }
        __syncthreads();

        // ---- Phase B: online softmax (8 lanes per row) ----
        {
            const int r  = t >> 3;           // 0..63
            const int c0 = (t & 7) << 2;     // 0,4,...,28
            float v0 = Spart[0][r][c0 + 0] + Spart[1][r][c0 + 0];
            float v1 = Spart[0][r][c0 + 1] + Spart[1][r][c0 + 1];
            float v2 = Spart[0][r][c0 + 2] + Spart[1][r][c0 + 2];
            float v3 = Spart[0][r][c0 + 3] + Spart[1][r][c0 + 3];
            float mx = fmaxf(fmaxf(v0, v1), fmaxf(v2, v3));
            mx = fmaxf(mx, __shfl_xor(mx, 1));
            mx = fmaxf(mx, __shfl_xor(mx, 2));
            mx = fmaxf(mx, __shfl_xor(mx, 4));
            const float mold = sm[r], lold = sl[r];
            const float mnew = fmaxf(mold, mx);
            const float al   = __expf(mold - mnew);
            float p0 = __expf(v0 - mnew);
            float p1 = __expf(v1 - mnew);
            float p2 = __expf(v2 - mnew);
            float p3 = __expf(v3 - mnew);
            Pc[r][c0 + 0] = (bh)p0;
            Pc[r][c0 + 1] = (bh)p1;
            Pc[r][c0 + 2] = (bh)p2;
            Pc[r][c0 + 3] = (bh)p3;
            float sum = (p0 + p1) + (p2 + p3);
            sum += __shfl_xor(sum, 1);
            sum += __shfl_xor(sum, 2);
            sum += __shfl_xor(sum, 4);
            if ((t & 7) == 0) { sm[r] = mnew; sl[r] = lold * al + sum; salpha[r] = al; }
        }
        __syncthreads();

        // ---- Phase C: O = alpha*O + P @ Vt_chunk (16 WMMAs) ----
        {
            const int mb = (lane >> 4) << 3;
            float al[8];
#pragma unroll
            for (int r = 0; r < 8; ++r) al[r] = salpha[qt * 16 + mb + r];
            bh16 pA = load_a_bf16(&Pc[0][0], 32, qt * 16, 0);
#pragma unroll
            for (int nt = 0; nt < 16; ++nt) {
#pragma unroll
                for (int r = 0; r < 8; ++r) o[nt][r] *= al[r];
                bh16 b = load_b_bf16(Vt, 8192, dq * 256 + nt * 16, key0);
                o[nt] = wmma_bf16(pA, b, o[nt]);
            }
        }
        __syncthreads();
    }

    // ---- Epilogue: store UNnormalized partial O and per-row stats ----
    {
        float* Op = Opart + (size_t)s * 8192 * 1024;
        const int n  = lane & 15;
        const int mb = (lane >> 4) << 3;
#pragma unroll
        for (int nt = 0; nt < 16; ++nt) {
#pragma unroll
            for (int r = 0; r < 8; ++r)
                Op[(size_t)(q0 + qt * 16 + mb + r) * 1024 + dq * 256 + nt * 16 + n] =
                    o[nt][r];
        }
        if (t < 64) {
            Mst[(size_t)s * 8192 + q0 + t] = sm[t];
            Lst[(size_t)s * 8192 + q0 + t] = sl[t];
        }
    }
}

// ---------------------------------------------------------------------------
// Kernel 3: combine the two KV-split partials.
//   out[i,:] = (O0[i,:]*e^{m0-m} + O1[i,:]*e^{m1-m}) / (l0*e^{m0-m}+l1*e^{m1-m})
// ---------------------------------------------------------------------------
__global__ __launch_bounds__(256) void combine_kernel(
    const float* __restrict__ Opart, const float* __restrict__ Mst,
    const float* __restrict__ Lst, float* __restrict__ Out)
{
    const int row = blockIdx.x;     // 0..8191
    const float m0 = Mst[row], m1 = Mst[8192 + row];
    const float l0 = Lst[row], l1 = Lst[8192 + row];
    const float m  = fmaxf(m0, m1);
    const float w0 = __expf(m0 - m), w1 = __expf(m1 - m);
    const float linv = 1.0f / (l0 * w0 + l1 * w1);
    const int c = threadIdx.x * 4;  // 256 threads * 4 = 1024 cols
    vfloat4 x0 = *(const vfloat4*)(Opart + (size_t)row * 1024 + c);
    vfloat4 x1 = *(const vfloat4*)(Opart + (size_t)8192 * 1024 + (size_t)row * 1024 + c);
    vfloat4 y;
#pragma unroll
    for (int j = 0; j < 4; ++j) y[j] = (x0[j] * w0 + x1[j] * w1) * linv;
    *(vfloat4*)(Out + (size_t)row * 1024 + c) = y;
}

// ---------------------------------------------------------------------------
// Launcher
// ---------------------------------------------------------------------------
extern "C" void kernel_launch(void* const* d_in, const int* in_sizes, int n_in,
                              void* d_out, int out_size, void* d_ws, size_t ws_size,
                              hipStream_t stream) {
    (void)in_sizes; (void)n_in; (void)out_size; (void)ws_size;
    const float* query = (const float*)d_in[0];
    const float* key   = (const float*)d_in[1];
    const float* value = (const float*)d_in[2];
    const float* Wq    = (const float*)d_in[3];
    const float* bq    = (const float*)d_in[4];
    const float* Wk    = (const float*)d_in[5];
    const float* bk    = (const float*)d_in[6];
    const float* Wv    = (const float*)d_in[7];
    const float* bv    = (const float*)d_in[8];

    const size_t E  = (size_t)8192 * 1024;   // elements per activation matrix
    const size_t WE = (size_t)1024 * 1024;   // elements per weight matrix

    // Workspace layout (bf16 region, then f32 region). ~130 MB total.
    bh* Qb   = (bh*)d_ws;      // projected Q (pre-scaled by 1/sqrt(1024))
    bh* Kb   = Qb + E;
    bh* Vtb  = Kb + E;         // projected V, transposed (1024 x 8192)
    bh* Xbuf = Vtb + E;        // reusable converted-input buffer
    bh* Wbuf = Xbuf + E;       // reusable converted-weight buffer
    float* Opart = (float*)(Wbuf + WE);       // [2][8192][1024] partial O
    float* MstP  = Opart + 2 * E;             // [2][8192]
    float* LstP  = MstP + 2 * 8192;           // [2][8192]

    const dim3 cvb(256);
    const dim3 cvgX((unsigned)(E / (256 * 8)));   // 4096 blocks, activations
    const dim3 cvgW((unsigned)(WE / (256 * 8)));  // 512 blocks, weights
    const dim3 pg(32, 8), pb(256);

    // Q projection (folds the 1/sqrt(dk) score scale into Q)
    cvt_bf16_kernel<<<cvgX, cvb, 0, stream>>>(query, Xbuf, (int)E);
    cvt_bf16_kernel<<<cvgW, cvb, 0, stream>>>(Wq, Wbuf, (int)WE);
    proj_gemm_kernel<<<pg, pb, 0, stream>>>(Xbuf, Wbuf, bq, Qb, 1.0f / 32.0f, 0);
    // K projection
    cvt_bf16_kernel<<<cvgX, cvb, 0, stream>>>(key, Xbuf, (int)E);
    cvt_bf16_kernel<<<cvgW, cvb, 0, stream>>>(Wk, Wbuf, (int)WE);
    proj_gemm_kernel<<<pg, pb, 0, stream>>>(Xbuf, Wbuf, bk, Kb, 1.0f, 0);
    // V projection, stored transposed
    cvt_bf16_kernel<<<cvgX, cvb, 0, stream>>>(value, Xbuf, (int)E);
    cvt_bf16_kernel<<<cvgW, cvb, 0, stream>>>(Wv, Wbuf, (int)WE);
    proj_gemm_kernel<<<pg, pb, 0, stream>>>(Xbuf, Wbuf, bv, Vtb, 1.0f, 1);

    // Flash attention, KV-split = 2, then combine.
    flash_attn_kernel<<<dim3(128, 2), dim3(512), 0, stream>>>(
        Qb, Kb, Vtb, Opart, MstP, LstP);
    combine_kernel<<<dim3(8192), dim3(256), 0, stream>>>(
        Opart, MstP, LstP, (float*)d_out);
}
